// Gemma4TextAttention_154618823077
// MI455X (gfx1250) — matrix operand order repaired and checked by reading, not compiled, not run on hardware
//
#include <hip/hip_runtime.h>
#include <cstdint>
#include <cstddef>

typedef __bf16 bf16;
typedef __bf16 v16bf __attribute__((ext_vector_type(16)));
typedef __bf16 v8bf  __attribute__((ext_vector_type(8)));
typedef float  v8f   __attribute__((ext_vector_type(8)));
typedef int    v4i   __attribute__((ext_vector_type(4)));

typedef __attribute__((address_space(1))) v4i gv4i;   // global <4 x i32>
typedef __attribute__((address_space(3))) v4i lv4i;   // LDS    <4 x i32>

#define B_   2
#define T_   2048
#define H_   8
#define KH_  2
#define DH_  256
#define HID_ 2048
#define WIN_ 512
#define EPS_ 1e-6f

__device__ __forceinline__ bf16 f2bf(float f) {
  uint32_t u = __builtin_bit_cast(uint32_t, f);
  u += 0x7FFFu + ((u >> 16) & 1u);           // round-to-nearest-even
  uint16_t h = (uint16_t)(u >> 16);
  return __builtin_bit_cast(bf16, h);
}

// 16B global -> LDS DMA (no VGPR round-trip), tracked by ASYNCcnt.
__device__ __forceinline__ void async_cp_b128(void* lds, const void* g) {
  __builtin_amdgcn_global_load_async_to_lds_b128((gv4i*)g, (lv4i*)lds, 0, 0);
}
__device__ __forceinline__ void wait_async0() {
  asm volatile("s_wait_asynccnt 0" ::: "memory");
}

// ---------------------------------------------------------------- fp32->bf16
__global__ void cvt_f32_bf16(const float* __restrict__ in, bf16* __restrict__ out, int n) {
  for (int i = blockIdx.x * blockDim.x + threadIdx.x; i < n; i += gridDim.x * blockDim.x)
    out[i] = f2bf(in[i]);
}

// ----------------------------------------------- C[M,N] = A[M,K] * W[N,K]^T
// bf16 in, fp32 out. 64x64 block tile, 8 waves, BK=32, double-buffered
// async global->LDS staging pipelined against WMMA.
__global__ __launch_bounds__(256) void gemm_bf16(
    const bf16* __restrict__ A, const bf16* __restrict__ W,
    float* __restrict__ C, int M, int N, int K) {
  __shared__ __align__(16) bf16 As[2][64][40];   // +8 pad -> 20-bank row shift
  __shared__ __align__(16) bf16 Bs[2][64][40];

  const int tid  = threadIdx.x;
  const int lane = tid & 31;
  const int w    = tid >> 5;
  const int m0   = blockIdx.y * 64;
  const int n0   = blockIdx.x * 64;
  const int mo   = 16 * (w & 3);
  const int no   = 32 * (w >> 2);
  const int lrow = tid >> 2;
  const int lcol = (tid & 3) * 8;
  const int hl   = lane >> 4;       // half-wave
  const int lm   = lane & 15;

  const bf16* ag = &A[(size_t)(m0 + lrow) * K + lcol];
  const bf16* wg = &W[(size_t)(n0 + lrow) * K + lcol];

  v8f acc0 = {0.f,0.f,0.f,0.f,0.f,0.f,0.f,0.f};
  v8f acc1 = {0.f,0.f,0.f,0.f,0.f,0.f,0.f,0.f};

  // prologue: DMA tile 0 into buffer 0
  async_cp_b128(&As[0][lrow][lcol], ag);
  async_cp_b128(&Bs[0][lrow][lcol], wg);

  const int nk = K / 32;
  for (int t = 0; t < nk; ++t) {
    const int cur = t & 1;
    wait_async0();        // my copies for buf[cur] have landed
    __syncthreads();      // everyone's copies landed; buf[1-cur] free
    if (t + 1 < nk) {     // overlap next tile DMA with this tile's WMMA
      async_cp_b128(&As[1 - cur][lrow][lcol], ag + (size_t)(t + 1) * 32);
      async_cp_b128(&Bs[1 - cur][lrow][lcol], wg + (size_t)(t + 1) * 32);
    }

    // A fragment: lane m = lm, K = hl*16 .. +15
    v16bf af;
    {
      const bf16* p = &As[cur][mo + lm][hl * 16];
      #pragma unroll
      for (int i = 0; i < 8; ++i) af[i] = p[i];
      #pragma unroll
      for (int i = 0; i < 8; ++i) af[8 + i] = p[8 + i];
    }
    // B fragments: lane n = lm, K = hl*16 .. +15
    v16bf bf0, bf1;
    {
      const bf16* p = &Bs[cur][no + lm][hl * 16];
      #pragma unroll
      for (int i = 0; i < 16; ++i) bf0[i] = p[i];
      const bf16* q = &Bs[cur][no + 16 + lm][hl * 16];
      #pragma unroll
      for (int i = 0; i < 16; ++i) bf1[i] = q[i];
    }
    acc0 = __builtin_amdgcn_wmma_f32_16x16x32_bf16(false, af, false, bf0, (short)0, acc0, false, false);
    acc1 = __builtin_amdgcn_wmma_f32_16x16x32_bf16(false, af, false, bf1, (short)0, acc1, false, false);
  }

  #pragma unroll
  for (int r = 0; r < 8; ++r) {
    const int row = m0 + mo + r + 8 * hl;
    C[(size_t)row * N + n0 + no + lm]      = acc0[r];
    C[(size_t)row * N + n0 + no + 16 + lm] = acc1[r];
  }
}

// ------------------------------------ per-head RMSNorm (+optional scale) + RoPE
// in:  (B,T,nH,DH) fp32   out: (B,nH,T,DH) bf16
__global__ __launch_bounds__(256) void norm_rope_kernel(
    const float* __restrict__ in, const float* __restrict__ normw,
    const float* __restrict__ cosT, const float* __restrict__ sinT,
    bf16* __restrict__ out, int nH, int doRope) {
  const int h = blockIdx.x, t = blockIdx.y, b = blockIdx.z;
  const int d = threadIdx.x;
  const size_t base = ((size_t)(b * T_ + t) * nH + h) * DH_;

  float x = in[base + d];
  float v = x * x;
  #pragma unroll
  for (int off = 16; off > 0; off >>= 1) v += __shfl_xor(v, off);
  __shared__ float red[8];
  if ((threadIdx.x & 31) == 0) red[threadIdx.x >> 5] = v;
  __syncthreads();
  float ms = 0.f;
  #pragma unroll
  for (int i = 0; i < 8; ++i) ms += red[i];
  ms = ms / (float)DH_ + EPS_;
  const float inv = rsqrtf(ms);

  float y = x * inv;
  if (normw) y *= normw[d];
  if (doRope) {
    const int dp = (d < DH_ / 2) ? d + DH_ / 2 : d - DH_ / 2;
    float o = in[base + dp] * inv;
    if (normw) o *= normw[dp];
    const float rot = (d < DH_ / 2) ? -o : o;
    const size_t ci = (size_t)(b * T_ + t) * DH_ + d;
    y = y * cosT[ci] + rot * sinT[ci];
  }
  out[((size_t)(b * nH + h) * T_ + t) * DH_ + d] = f2bf(y);
}

// ------------------------------------------------ flash attention (windowed)
// Q:(B,H,T,DH)  K,V:(B,KH,T,DH)  O:(B,T,H*DH)  bf16 data, f32 accumulation.
// 4 waves/block; wave wq owns query rows [q0+16*wq, +16). 32-key tiles.
// K tile staged via async DMA (overlapped with manual V transpose).
__global__ __launch_bounds__(128) void flash_attn_kernel(
    const bf16* __restrict__ Q, const bf16* __restrict__ Km,
    const bf16* __restrict__ Vm, bf16* __restrict__ O) {
  __shared__ __align__(16) bf16 Ks[32][264];   // [key][d], pad 8
  __shared__ __align__(16) bf16 Vs[256][40];   // [d][key] transposed, pad 8
  __shared__ __align__(16) bf16 Ps[4][16][40]; // per-wave P transpose staging

  const int q0  = blockIdx.x * 64;
  const int h   = blockIdx.y;
  const int b   = blockIdx.z;
  const int kvh = h / (H_ / KH_);
  const int tid = threadIdx.x;
  const int lane = tid & 31;
  const int wq  = tid >> 5;
  const int hl  = lane >> 4;
  const int lm  = lane & 15;

  const bf16* kbase = Km + (size_t)(b * KH_ + kvh) * T_ * DH_;
  const bf16* vbase = Vm + (size_t)(b * KH_ + kvh) * T_ * DH_;

  // Q fragments held in registers for the whole kernel (8 x K=32 chunks)
  v16bf qf[8];
  {
    const int row = q0 + 16 * wq + lm;
    const bf16* qr = Q + ((size_t)((b * H_ + h) * T_) + row) * DH_;
    #pragma unroll
    for (int kk = 0; kk < 8; ++kk) {
      const int dbase = 32 * kk + hl * 16;
      #pragma unroll
      for (int i = 0; i < 8; ++i) qf[kk][i] = qr[dbase + i];
      #pragma unroll
      for (int i = 0; i < 8; ++i) qf[kk][8 + i] = qr[dbase + 8 + i];
    }
  }

  v8f oacc[16];
  #pragma unroll
  for (int i = 0; i < 16; ++i) oacc[i] = (v8f){0.f,0.f,0.f,0.f,0.f,0.f,0.f,0.f};
  float rmax[8], rsum[8];
  #pragma unroll
  for (int r = 0; r < 8; ++r) { rmax[r] = -3.0e28f; rsum[r] = 0.f; }

  int kt0 = q0 - (WIN_ - 1);
  if (kt0 < 0) kt0 = 0;
  kt0 &= ~31;
  const int kt1 = q0 + 64;

  for (int kt = kt0; kt < kt1; kt += 32) {
    __syncthreads();   // previous tile's compute done: LDS free for restaging
    // K tile: async DMA, row-major. V tile: manual transpose into Vs[d][k].
    #pragma unroll
    for (int it = 0; it < 8; ++it) {
      const int c    = tid + it * 128;       // 0..1023
      const int krow = c >> 5;
      const int dc   = (c & 31) * 8;
      async_cp_b128(&Ks[krow][dc], &kbase[(size_t)(kt + krow) * DH_ + dc]);
      const v8bf vv = *(const v8bf*)&vbase[(size_t)(kt + krow) * DH_ + dc];
      #pragma unroll
      for (int i = 0; i < 8; ++i) Vs[dc + i][krow] = vv[i];
    }
    if (kt + 32 < kt1)
      __builtin_prefetch(&vbase[(size_t)(kt + 32) * DH_ + (size_t)lane * 64], 0, 1);
    wait_async0();
    __syncthreads();

    // S = Q K^T : two 16-key subtiles, contraction over DH in 8 steps
    v8f s0 = (v8f){0.f,0.f,0.f,0.f,0.f,0.f,0.f,0.f};
    v8f s1 = (v8f){0.f,0.f,0.f,0.f,0.f,0.f,0.f,0.f};
    #pragma unroll
    for (int kk = 0; kk < 8; ++kk) {
      v16bf b0, b1;
      const bf16* p0 = &Ks[lm][32 * kk + hl * 16];
      const bf16* p1 = &Ks[16 + lm][32 * kk + hl * 16];
      #pragma unroll
      for (int i = 0; i < 16; ++i) b0[i] = p0[i];
      #pragma unroll
      for (int i = 0; i < 16; ++i) b1[i] = p1[i];
      s0 = __builtin_amdgcn_wmma_f32_16x16x32_bf16(false, qf[kk], false, b0, (short)0, s0, false, false);
      s1 = __builtin_amdgcn_wmma_f32_16x16x32_bf16(false, qf[kk], false, b1, (short)0, s1, false, false);
    }

    // causal + sliding-window mask, online softmax (row m = r + 8*hl)
    #pragma unroll
    for (int r = 0; r < 8; ++r) {
      const int qi = q0 + 16 * wq + r + 8 * hl;
      const int j0 = kt + lm;
      const int j1 = kt + 16 + lm;
      float v0 = (j0 <= qi && (qi - j0) < WIN_) ? s0[r] : -1.0e30f;
      float v1 = (j1 <= qi && (qi - j1) < WIN_) ? s1[r] : -1.0e30f;
      float tmax = fmaxf(v0, v1);
      #pragma unroll
      for (int off = 8; off > 0; off >>= 1) tmax = fmaxf(tmax, __shfl_xor(tmax, off));
      const float nm    = fmaxf(rmax[r], tmax);
      const float scale = __expf(rmax[r] - nm);
      rmax[r] = nm;
      const float p0 = __expf(v0 - nm);
      const float p1 = __expf(v1 - nm);
      float ps = p0 + p1;
      #pragma unroll
      for (int off = 8; off > 0; off >>= 1) ps += __shfl_xor(ps, off);
      rsum[r] = rsum[r] * scale + ps;
      #pragma unroll
      for (int dt = 0; dt < 16; ++dt) oacc[dt][r] *= scale;
      const int prow = r + 8 * hl;
      Ps[wq][prow][lm]      = f2bf(p0);
      Ps[wq][prow][16 + lm] = f2bf(p1);
    }

    // P in A-fragment layout (transpose via LDS; in-order DS within wave)
    v16bf pf;
    {
      const bf16* p = &Ps[wq][lm][hl * 16];
      #pragma unroll
      for (int i = 0; i < 8; ++i) pf[i] = p[i];
      #pragma unroll
      for (int i = 0; i < 8; ++i) pf[8 + i] = p[8 + i];
    }
    // O += P * V over all 16 d-subtiles
    #pragma unroll
    for (int dt = 0; dt < 16; ++dt) {
      v16bf vf;
      const bf16* p = &Vs[16 * dt + lm][hl * 16];
      #pragma unroll
      for (int i = 0; i < 16; ++i) vf[i] = p[i];
      oacc[dt] = __builtin_amdgcn_wmma_f32_16x16x32_bf16(false, pf, false, vf, (short)0, oacc[dt], false, false);
    }
  }

  // epilogue: normalize and store to (B,T,H*DH) bf16
  #pragma unroll
  for (int r = 0; r < 8; ++r) {
    const int row = q0 + 16 * wq + r + 8 * hl;
    const float is = 1.0f / rsum[r];
    bf16* orow = O + (size_t)(b * T_ + row) * (H_ * DH_) + h * DH_;
    #pragma unroll
    for (int dt = 0; dt < 16; ++dt)
      orow[16 * dt + lm] = f2bf(oacc[dt][r] * is);
  }
}

// ---------------------------------------------------------------- launcher
extern "C" void kernel_launch(void* const* d_in, const int* in_sizes, int n_in,
                              void* d_out, int out_size, void* d_ws, size_t ws_size,
                              hipStream_t stream) {
  (void)in_sizes; (void)n_in; (void)out_size; (void)ws_size;
  const float* hs   = (const float*)d_in[0];
  const float* Wq   = (const float*)d_in[1];
  const float* Wk   = (const float*)d_in[2];
  const float* Wv   = (const float*)d_in[3];
  const float* Wo   = (const float*)d_in[4];
  const float* qnw  = (const float*)d_in[5];
  const float* knw  = (const float*)d_in[6];
  const float* cosT = (const float*)d_in[7];
  const float* sinT = (const float*)d_in[8];
  // d_in[9] (attention_mask) reproduced analytically in-kernel.

  char* ws = (char*)d_ws;
  size_t off = 0;
  auto alloc = [&](size_t bytes) -> void* {
    void* p = ws + off;
    off = (off + bytes + 255) & ~(size_t)255;
    return p;
  };
  const size_t MT = (size_t)B_ * T_;                       // 4096 rows
  bf16*  hbf  = (bf16*)alloc(MT * HID_ * 2);
  bf16*  wqbf = (bf16*)alloc((size_t)H_ * DH_ * HID_ * 2);
  bf16*  wkbf = (bf16*)alloc((size_t)KH_ * DH_ * HID_ * 2);
  bf16*  wvbf = (bf16*)alloc((size_t)KH_ * DH_ * HID_ * 2);
  bf16*  wobf = (bf16*)alloc((size_t)HID_ * H_ * DH_ * 2);
  float* qf   = (float*)alloc(MT * H_ * DH_ * 4);
  float* kf   = (float*)alloc(MT * KH_ * DH_ * 4);
  float* vf   = (float*)alloc(MT * KH_ * DH_ * 4);
  bf16*  qbf  = (bf16*)alloc(MT * H_ * DH_ * 2);
  bf16*  kbf2 = (bf16*)alloc(MT * KH_ * DH_ * 2);
  bf16*  vbf2 = (bf16*)alloc(MT * KH_ * DH_ * 2);
  bf16*  abf  = (bf16*)qf;   // reuse q fp32 region (dead after norm/rope)

  cvt_f32_bf16<<<1024, 256, 0, stream>>>(hs, hbf, (int)(MT * HID_));
  cvt_f32_bf16<<<512,  256, 0, stream>>>(Wq, wqbf, H_ * DH_ * HID_);
  cvt_f32_bf16<<<256,  256, 0, stream>>>(Wk, wkbf, KH_ * DH_ * HID_);
  cvt_f32_bf16<<<256,  256, 0, stream>>>(Wv, wvbf, KH_ * DH_ * HID_);
  cvt_f32_bf16<<<512,  256, 0, stream>>>(Wo, wobf, HID_ * H_ * DH_);

  gemm_bf16<<<dim3((H_ * DH_) / 64, MT / 64), 256, 0, stream>>>(hbf, wqbf, qf, (int)MT, H_ * DH_, HID_);
  gemm_bf16<<<dim3((KH_ * DH_) / 64, MT / 64), 256, 0, stream>>>(hbf, wkbf, kf, (int)MT, KH_ * DH_, HID_);
  gemm_bf16<<<dim3((KH_ * DH_) / 64, MT / 64), 256, 0, stream>>>(hbf, wvbf, vf, (int)MT, KH_ * DH_, HID_);

  norm_rope_kernel<<<dim3(H_,  T_, B_), 256, 0, stream>>>(qf, qnw, cosT, sinT, qbf,  H_,  1);
  norm_rope_kernel<<<dim3(KH_, T_, B_), 256, 0, stream>>>(kf, knw, cosT, sinT, kbf2, KH_, 1);
  norm_rope_kernel<<<dim3(KH_, T_, B_), 256, 0, stream>>>(vf, nullptr, cosT, sinT, vbf2, KH_, 0);

  flash_attn_kernel<<<dim3(T_ / 64, H_, B_), 128, 0, stream>>>(qbf, kbf2, vbf2, abf);

  gemm_bf16<<<dim3(HID_ / 64, MT / 64), 256, 0, stream>>>(abf, wobf, (float*)d_out, (int)MT, HID_, H_ * DH_);
}
